// Attention_6167573037833
// MI455X (gfx1250) — compile-verified, matrix operand order checked by reading, not run on hardware
//
#include <hip/hip_runtime.h>
#include <hip/hip_bf16.h>

// ---------------- CDNA5 WMMA types ----------------
typedef __attribute__((ext_vector_type(16))) _Float16 v16h;
typedef __attribute__((ext_vector_type(8)))  _Float16 v8h;
typedef __attribute__((ext_vector_type(8)))  float    v8f;

union Frag16 { v16h v; v8h h[2]; };

// Dimensions (compile-time)
#define BB 4
#define SS 2048
#define DD 1024
#define HH 16
#define HD 64
#define MM (BB * SS)   // 8192

// A/B fragment loader for v_wmma_f32_16x16x32_f16.
// ISA 16-bit A layout: lane L (0..15) holds row M=L; VGPR0..3 = K (kb..kb+7),
// VGPR4..7 = K (kb+16..kb+23), where kb = (L>=16 ? 8 : 0). B is symmetric on
// columns (implicit B = rows of the "Bt" array => computes A * Bt^T).
__device__ __forceinline__ v16h ld_frag(const _Float16* __restrict__ base,
                                        int row0, int k0, int ld, int lane) {
  const int r  = row0 + (lane & 15);
  const int kb = k0 + ((lane >> 4) << 3);
  const _Float16* p = base + (size_t)r * ld + kb;
  Frag16 f;
  f.h[0] = *(const v8h*)(p);        // K = kb .. kb+7
  f.h[1] = *(const v8h*)(p + 16);   // K = kb+16 .. kb+23
  return f.v;
}

__device__ __forceinline__ v8f wmma16(v16h a, v16h b, v8f c) {
  // (neg_a, A, neg_b, B, c_mod, C, reuse_a, reuse_b)
  return __builtin_amdgcn_wmma_f32_16x16x32_f16(false, a, false, b, (short)0, c,
                                                false, false);
}

// ---------------- conversion kernels ----------------
__global__ __launch_bounds__(256) void cvt_f32_to_f16(const float* __restrict__ in,
                                                      _Float16* __restrict__ out,
                                                      int n) {
  int i = blockIdx.x * 256 + threadIdx.x;
  if (i < n) out[i] = (_Float16)in[i];
}

// Wt[n*DD + k] = (f16) W[k*DD + n]  (coalesced writes)
__global__ __launch_bounds__(256) void cvt_transpose(const float* __restrict__ W,
                                                     _Float16* __restrict__ Wt) {
  int idx = blockIdx.x * 256 + threadIdx.x;  // over DD*DD, k fastest
  int n = idx >> 10;
  int k = idx & 1023;
  Wt[idx] = (_Float16)W[(size_t)k * DD + n];
}

// ---------------- generic projection GEMM ----------------
// C[M,N] = A[M,K](f16) * Bt[N,K](f16)^T + bias
// MODE 0: fp32 out, row-major [M,N]
// MODE 1: f16 out, head layout [B,H,S,HD]   (Q, K)
// MODE 2: f16 out, transposed  [B,H,HD,S]   (V)
template <int MODE>
__global__ __launch_bounds__(256) void gemm_proj(const _Float16* __restrict__ A,
                                                 const _Float16* __restrict__ Bt,
                                                 const float* __restrict__ bias,
                                                 void* __restrict__ out,
                                                 int Ndim, int Kdim) {
  const int lane = threadIdx.x & 31;
  const int wave = threadIdx.x >> 5;
  const int m0 = blockIdx.x * 128 + (wave >> 2) * 64;  // 4 M-tiles per wave
  const int n0 = blockIdx.y * 64 + (wave & 3) * 16;

  v8f acc[4] = {};
  for (int k0 = 0; k0 < Kdim; k0 += 32) {
    v16h bf = ld_frag(Bt, n0, k0, Kdim, lane);
#pragma unroll
    for (int t = 0; t < 4; ++t) {
      v16h af = ld_frag(A, m0 + 16 * t, k0, Kdim, lane);
      acc[t] = wmma16(af, bf, acc[t]);
    }
  }

  const int n = n0 + (lane & 15);
  const float bn = bias[n];
#pragma unroll
  for (int t = 0; t < 4; ++t) {
#pragma unroll
    for (int r = 0; r < 8; ++r) {
      const int m = m0 + 16 * t + r + ((lane >> 4) << 3);
      const float v = acc[t][r] + bn;
      if (MODE == 0) {
        ((float*)out)[(size_t)m * Ndim + n] = v;
      } else {
        const int b = m >> 11, s = m & (SS - 1);   // S = 2048
        const int h = n >> 6, hd = n & (HD - 1);   // HD = 64
        if (MODE == 1)
          ((_Float16*)out)[(((size_t)(b * HH + h)) * SS + s) * HD + hd] = (_Float16)v;
        else
          ((_Float16*)out)[(((size_t)(b * HH + h)) * HD + hd) * SS + s] = (_Float16)v;
      }
    }
  }
}

// ---------------- fused scores -> softmax -> probs + context ----------------
// One wave owns 16 query rows. Q fragments live in registers for the whole
// kernel. Sweep 1 recomputes QK^T tiles and keeps per-lane online (max,sum);
// a single 4-step shfl_xor combine yields exact row stats. Sweep 2 recomputes
// the tiles, writes normalized fp32 probs (the only 1.07GB that must touch
// HBM), stages each 16x32 prob chunk to LDS as f16, and reloads it as a WMMA
// A-fragment to accumulate P@V. Scores are never spilled to memory.
__global__ __launch_bounds__(256) void fused_attn(const _Float16* __restrict__ Qh,
                                                  const _Float16* __restrict__ Kh,
                                                  const _Float16* __restrict__ Vt,
                                                  const float* __restrict__ mask,
                                                  float* __restrict__ probs,
                                                  _Float16* __restrict__ ctx) {
  __shared__ _Float16 ldsP[8][16 * 32];  // per-wave 16x32 f16 staging tile (8KB)
  const int lane = threadIdx.x & 31;
  const int wave = threadIdx.x >> 5;
  const int bh = blockIdx.z;
  const int b = bh >> 4, h = bh & 15;
  const int q0 = blockIdx.x * 128 + wave * 16;

  const _Float16* Qb = Qh + (size_t)bh * SS * HD;
  const _Float16* Kb = Kh + (size_t)bh * SS * HD;
  const _Float16* Vb = Vt + (size_t)bh * HD * SS;
  float* Pout = probs + (size_t)bh * SS * SS;
  const float* mrow = mask + (size_t)b * SS;

  const float scale = 0.125f;        // 1/sqrt(64)
  const int nc = lane & 15;          // column within 16-wide tile
  const int rg = (lane >> 4) << 3;   // row offset of this half-wave (0 or 8)

  const v16h qf0 = ld_frag(Qb, q0, 0, HD, lane);
  const v16h qf1 = ld_frag(Qb, q0, 32, HD, lane);

  // ---- Sweep 1: per-lane online softmax stats over 128 key tiles ----
  float mx[8], lsum[8];
#pragma unroll
  for (int r = 0; r < 8; ++r) { mx[r] = -3.0e38f; lsum[r] = 0.f; }

  for (int j = 0; j < 128; ++j) {
    v8f s = {};
    s = wmma16(qf0, ld_frag(Kb, j * 16, 0, HD, lane), s);
    s = wmma16(qf1, ld_frag(Kb, j * 16, 32, HD, lane), s);
    const float mk = mrow[j * 16 + nc] * -1e9f;
#pragma unroll
    for (int r = 0; r < 8; ++r) {
      const float x = s[r] * scale + mk;
      const float mn = fmaxf(mx[r], x);
      lsum[r] = lsum[r] * __expf(mx[r] - mn) + __expf(x - mn);
      mx[r] = mn;
    }
  }
  // combine (max,sum) across the 16 lanes sharing each row
#pragma unroll
  for (int off = 1; off < 16; off <<= 1) {
#pragma unroll
    for (int r = 0; r < 8; ++r) {
      const float mo = __shfl_xor(mx[r], off);
      const float lo = __shfl_xor(lsum[r], off);
      const float mn = fmaxf(mx[r], mo);
      lsum[r] = lsum[r] * __expf(mx[r] - mn) + lo * __expf(mo - mn);
      mx[r] = mn;
    }
  }
  float invl[8];
#pragma unroll
  for (int r = 0; r < 8; ++r) invl[r] = 1.0f / lsum[r];

  // ---- Sweep 2: recompute, emit normalized probs, accumulate context ----
  v8f cacc[4] = {};
  for (int c = 0; c < 64; ++c) {  // 32-column chunks (2 key tiles)
#pragma unroll
    for (int half = 0; half < 2; ++half) {
      const int j = 2 * c + half;
      v8f s = {};
      s = wmma16(qf0, ld_frag(Kb, j * 16, 0, HD, lane), s);
      s = wmma16(qf1, ld_frag(Kb, j * 16, 32, HD, lane), s);
      const float mk = mrow[j * 16 + nc] * -1e9f;
#pragma unroll
      for (int r = 0; r < 8; ++r) {
        const float p = __expf(s[r] * scale + mk - mx[r]) * invl[r];
        Pout[(size_t)(q0 + r + rg) * SS + j * 16 + nc] = p;            // fp32 out
        ldsP[wave][(r + rg) * 32 + half * 16 + nc] = (_Float16)p;      // stage f16
      }
    }
    // Same-wave DS ops execute in order; make the dependency explicit anyway.
    asm volatile("s_wait_dscnt 0" ::: "memory");
    const v16h pf = ld_frag(&ldsP[wave][0], 0, 0, 32, lane);
#pragma unroll
    for (int t = 0; t < 4; ++t)
      cacc[t] = wmma16(pf, ld_frag(Vb, t * 16, c * 32, SS, lane), cacc[t]);
  }

  // heads merged on store: ctx[b][s][h*HD+hd], f16
#pragma unroll
  for (int t = 0; t < 4; ++t) {
    const int hd = 16 * t + nc;
#pragma unroll
    for (int r = 0; r < 8; ++r) {
      const int srow = q0 + r + rg;
      ctx[((size_t)(b * SS + srow)) * DD + h * HD + hd] = (_Float16)cacc[t][r];
    }
  }
}

// ---------------- host launcher ----------------
extern "C" void kernel_launch(void* const* d_in, const int* in_sizes, int n_in,
                              void* d_out, int out_size, void* d_ws, size_t ws_size,
                              hipStream_t stream) {
  (void)in_sizes; (void)n_in; (void)out_size; (void)ws_size;

  const float* q    = (const float*)d_in[0];
  const float* mask = (const float*)d_in[1];
  const float* Wq = (const float*)d_in[2]; const float* bq = (const float*)d_in[3];
  const float* Wk = (const float*)d_in[4]; const float* bk = (const float*)d_in[5];
  const float* Wv = (const float*)d_in[6]; const float* bv = (const float*)d_in[7];
  const float* Wo = (const float*)d_in[8]; const float* bo = (const float*)d_in[9];

  float* outc  = (float*)d_out;                     // context [B,S,D]
  float* attnw = outc + (size_t)MM * DD;            // probs   [B,H,S,S]

  // workspace layout (f16 elements)
  _Float16* ws  = (_Float16*)d_ws;
  _Float16* qh  = ws;                               // [M,D]
  _Float16* WtQ = qh  + (size_t)MM * DD;            // [D,D] each
  _Float16* WtK = WtQ + (size_t)DD * DD;
  _Float16* WtV = WtK + (size_t)DD * DD;
  _Float16* WtO = WtV + (size_t)DD * DD;
  _Float16* Qh  = WtO + (size_t)DD * DD;            // [B,H,S,HD]
  _Float16* Kh  = Qh  + (size_t)MM * DD;
  _Float16* Vt  = Kh  + (size_t)MM * DD;            // [B,H,HD,S]
  _Float16* ctx = Vt  + (size_t)MM * DD;            // [B,S,D]

  // Phase 0: precision conversion + weight transposes
  cvt_f32_to_f16<<<(MM * DD) / 256, 256, 0, stream>>>(q, qh, MM * DD);
  cvt_transpose<<<(DD * DD) / 256, 256, 0, stream>>>(Wq, WtQ);
  cvt_transpose<<<(DD * DD) / 256, 256, 0, stream>>>(Wk, WtK);
  cvt_transpose<<<(DD * DD) / 256, 256, 0, stream>>>(Wv, WtV);
  cvt_transpose<<<(DD * DD) / 256, 256, 0, stream>>>(Wo, WtO);

  // Phase 1: QKV projections (WMMA)
  dim3 gproj(MM / 128, DD / 64);
  gemm_proj<1><<<gproj, 256, 0, stream>>>(qh, WtQ, bq, Qh, DD, DD);
  gemm_proj<1><<<gproj, 256, 0, stream>>>(qh, WtK, bk, Kh, DD, DD);
  gemm_proj<2><<<gproj, 256, 0, stream>>>(qh, WtV, bv, Vt, DD, DD);

  // Phase 2: fused scores -> softmax -> probs(out) + context (score recompute,
  // no score spill; only the mandatory probs tensor touches HBM)
  fused_attn<<<dim3(SS / 128, 1, BB * HH), 256, 0, stream>>>(Qh, Kh, Vt, mask,
                                                             attnw, ctx);

  // Phase 3: output projection (fp32 epilogue)
  gemm_proj<0><<<gproj, 256, 0, stream>>>(ctx, WtO, bo, outc, DD, DD);
}
